// SelfAttention_36644660969992
// MI455X (gfx1250) — matrix operand order repaired and checked by reading, not compiled, not run on hardware
//
#include <hip/hip_runtime.h>
#include <hip/hip_bf16.h>
#include <cstdint>

#define SEQL 2048
#define NB   4
#define NH   16
#define HD   64
#define ED   1024

typedef __bf16 bf16_t;
typedef __attribute__((ext_vector_type(16))) __bf16 v16bf;
typedef __attribute__((ext_vector_type(8)))  __bf16 v8bf;
typedef __attribute__((ext_vector_type(8)))  float  v8f;

#define LOG2E 1.4426950408889634f

__device__ __forceinline__ v8f wmma_bf16(v16bf a, v16bf b, v8f c) {
    return __builtin_amdgcn_wmma_f32_16x16x32_bf16(
        /*neg_a=*/false, a, /*neg_b=*/false, b,
        /*c_mod=*/(short)0, c, /*reuse_a=*/false, /*reuse_b=*/false);
}

// A-fragment (16x32 bf16): lane holds row M=lane%16; element e -> K = e + 16*half.
// 'row' must already point at (row, K=0) of this lane. Two contiguous 16B chunks.
__device__ __forceinline__ v16bf load_a_bf16(const bf16_t* row, int half) {
    v16bf a;
    const bf16_t* p0 = row + 16 * half;
    const bf16_t* p1 = row + 16 * half + 8;
#pragma unroll
    for (int e = 0; e < 8; ++e) a[e] = p0[e];
#pragma unroll
    for (int e = 0; e < 8; ++e) a[8 + e] = p1[e];
    return a;
}

__device__ __forceinline__ v16bf load_a_f32(const float* row, int half) {
    v16bf a;
    const float* p0 = row + 16 * half;
    const float* p1 = row + 16 * half + 8;
#pragma unroll
    for (int e = 0; e < 8; ++e) a[e] = (bf16_t)p0[e];
#pragma unroll
    for (int e = 0; e < 8; ++e) a[8 + e] = (bf16_t)p1[e];
    return a;
}

// B-fragment (32x16 bf16): lane holds column N=lane%16; element e -> K = e + 16*half.
// 'p' must already point at (K = 16*half, this lane's column row in memory): 32B contiguous.
__device__ __forceinline__ v16bf load_b_bf16(const bf16_t* p) {
    v16bf b;
#pragma unroll
    for (int e = 0; e < 16; ++e) b[e] = p[e];
    return b;
}

__device__ __forceinline__ v16bf load_b_f32(const float* p) {
    v16bf b;
#pragma unroll
    for (int e = 0; e < 16; ++e) b[e] = (bf16_t)p[e];
    return b;
}

// Async DMA 16B per lane: global -> LDS, tracked by ASYNCcnt (CDNA5 path).
// lds_off is the wave-relative LDS byte offset (low 32 bits of a generic
// pointer to __shared__ memory per the flat-aperture mapping).
__device__ __forceinline__ void async_cp16(uint32_t lds_off, const bf16_t* g) {
    asm volatile("global_load_async_to_lds_b128 %0, %1, off"
                 :: "v"(lds_off), "v"(g) : "memory");
}
__device__ __forceinline__ uint32_t lds_off32(const void* p) {
    return (uint32_t)(uintptr_t)p;
}

// ---------------------------------------------------------------------------
// Kernel 0: Wo f32 -> bf16 (1M elements), so the big output GEMM reads bf16.
// ---------------------------------------------------------------------------
__global__ void wo_to_bf16(const float* __restrict__ Wo, bf16_t* __restrict__ Wob) {
    int i = blockIdx.x * blockDim.x + threadIdx.x;   // grid covers exactly E*E
    Wob[i] = (bf16_t)Wo[i];
}

// ---------------------------------------------------------------------------
// Kernel 1: QKV projections. One wave = 16 rows x 64 cols for one (matrix, n, h).
// q,k stored row-major bf16 [(n,h), seq, 64]; v stored TRANSPOSED [(n,h), 64, seq]
// (the WMMA C layout makes the transposed store a packed b128 per lane).
// ---------------------------------------------------------------------------
__global__ void qkv_proj(const float* __restrict__ V, const float* __restrict__ K,
                         const float* __restrict__ Q,
                         const float* __restrict__ Wv, const float* __restrict__ Wk,
                         const float* __restrict__ Wq,
                         bf16_t* __restrict__ qb, bf16_t* __restrict__ kb,
                         bf16_t* __restrict__ vtb) {
    int wave = threadIdx.x >> 5, lane = threadIdx.x & 31;
    int lo = lane & 15, half = lane >> 4;
    int g  = blockIdx.x * 8 + wave;        // 24576 waves
    int mt = g >> 13;                      // 0=v, 1=k, 2=q
    int r  = g & 8191;
    int n  = r >> 11;
    int h  = (r >> 7) & 15;
    int l0 = (r & 127) << 4;
    int nh = n * NH + h;

    const float* X = (mt == 0) ? V : (mt == 1) ? K : Q;
    const float* W = (mt == 0) ? Wv : (mt == 1) ? Wk : Wq;

    // A = X rows (f32 -> bf16), head slice: row stride E, base col h*64
    const float* arow = X + (size_t)(n * SEQL + l0 + lo) * ED + h * HD;
    v16bf A0 = load_a_f32(arow, half);        // d = 0..31
    v16bf A1 = load_a_f32(arow + 32, half);   // d = 32..63

#pragma unroll
    for (int nt = 0; nt < 4; ++nt) {
        // B = W^T tile: B[kk][nn] = W[nt*16+nn][kd0+kk]
        v16bf B0 = load_b_f32(W + (nt * 16 + lo) * HD + 16 * half);
        v16bf B1 = load_b_f32(W + (nt * 16 + lo) * HD + 32 + 16 * half);
        v8f c = {};
        c = wmma_bf16(A0, B0, c);
        c = wmma_bf16(A1, B1, c);

        if (mt == 0) {
            // transposed store: vtb[(nh*64 + d)*SEQ + l], C element i -> l = l0+8*half+i
            v8bf pk;
#pragma unroll
            for (int i = 0; i < 8; ++i) pk[i] = (bf16_t)c[i];
            *(v8bf*)(vtb + (size_t)(nh * HD + nt * 16 + lo) * SEQL + l0 + 8 * half) = pk;
        } else {
            bf16_t* dst = (mt == 1) ? kb : qb;
#pragma unroll
            for (int i = 0; i < 8; ++i)
                dst[(size_t)(nh * SEQL + l0 + i + 8 * half) * HD + nt * 16 + lo] = (bf16_t)c[i];
        }
    }
}

// ---------------------------------------------------------------------------
// Kernel 2: flash attention. Block = 8 waves = 128 q rows for one (n,h).
// K/V tiles are DMA'd into double-buffered LDS with global_load_async_to_lds
// (one 4KB tile per 256-thread block per instruction), overlapped with the
// current chunk's WMMAs and synchronized via s_wait_asynccnt + s_barrier.
// Per 32-wide K chunk: 4 WMMAs for S, online softmax (cross-lane reduce over
// the 16 lanes of each half-wave), P re-layout C->A via 1KB/wave LDS tile,
// then 4 WMMAs for P @ V (V transposed -> contiguous 32B ds reads).
// ---------------------------------------------------------------------------
__global__ void flash_attn(const bf16_t* __restrict__ qb, const bf16_t* __restrict__ kb,
                           const bf16_t* __restrict__ vtb, const int* __restrict__ mask,
                           bf16_t* __restrict__ ob) {
    __shared__ bf16_t Kl[2][32 * 64];   // K tile: 32 k-rows x 64 d (4KB each)
    __shared__ bf16_t Vl[2][64 * 32];   // V^T tile: 64 d-rows x 32 k (4KB each)
    __shared__ bf16_t Pl[8][16 * 32];   // per-wave P tile (1KB each)

    int tid  = threadIdx.x;
    int wave = tid >> 5, lane = tid & 31;
    int lo = lane & 15, half = lane >> 4;

    int b  = blockIdx.x;        // 1024 blocks
    int qt = b & 15;
    int nh = b >> 4;
    int n  = nh >> 4, h = nh & 15;
    int q0 = qt * 128 + wave * 16;

    const bf16_t* qbase = qb + (size_t)nh * SEQL * HD;
    const bf16_t* kbase = kb + (size_t)nh * SEQL * HD;
    const bf16_t* vbase = vtb + (size_t)nh * HD * SEQL;
    const int* mrow = mask + n * SEQL;

    // Per-thread 16B DMA slice coordinates (256 threads cover each 4KB tile)
    int krow = tid >> 3, kcol = (tid & 7) * 8;   // K tile: 32 rows x 64 elems
    int vrow = tid >> 2, vcol = (tid & 3) * 8;   // V tile: 64 rows x 32 elems

    const bf16_t* qrow = qbase + (size_t)(q0 + lo) * HD;
    v16bf A0 = load_a_bf16(qrow, half);
    v16bf A1 = load_a_bf16(qrow + 32, half);

    v8f o0 = {}, o1 = {}, o2 = {}, o3 = {};
    float rm[8], rl[8];
#pragma unroll
    for (int i = 0; i < 8; ++i) { rm[i] = -__builtin_inff(); rl[i] = 0.f; }

    const float SC = 0.03125f;   // 1/sqrt(1024)

    // Prologue: DMA chunk 0 into buffer 0 (ASYNCcnt = 2 per wave)
    async_cp16(lds_off32(&Kl[0][krow * 64 + kcol]), kbase + (size_t)krow * HD + kcol);
    async_cp16(lds_off32(&Vl[0][vrow * 32 + vcol]), vbase + (size_t)vrow * SEQL + vcol);

    for (int it = 0; it < SEQL / 32; ++it) {
        int k0  = it * 32;
        int cur = it & 1;
        int kn  = (k0 + 32) & (SEQL - 1);   // wraparound dummy prefetch on last iter

        // Prefetch next chunk into the other buffer (ASYNCcnt -> 4)
        async_cp16(lds_off32(&Kl[cur ^ 1][krow * 64 + kcol]),
                   kbase + (size_t)(kn + krow) * HD + kcol);
        async_cp16(lds_off32(&Vl[cur ^ 1][vrow * 32 + vcol]),
                   vbase + (size_t)vrow * SEQL + kn + vcol);

        // Async loads retire in order: <=2 outstanding means chunk `it` landed.
        asm volatile("s_wait_asynccnt 2" ::: "memory");
        __syncthreads();

        // S = Q @ K^T from LDS: two 16x16 tiles, 2 WMMAs each (d = 0..31, 32..63)
        v8f s0 = {}, s1 = {};
        {
            v16bf Bk;
            Bk = load_b_bf16(&Kl[cur][(lo)      * 64 + 16 * half]);
            s0 = wmma_bf16(A0, Bk, s0);
            Bk = load_b_bf16(&Kl[cur][(lo)      * 64 + 32 + 16 * half]);
            s0 = wmma_bf16(A1, Bk, s0);
            Bk = load_b_bf16(&Kl[cur][(16 + lo) * 64 + 16 * half]);
            s1 = wmma_bf16(A0, Bk, s1);
            Bk = load_b_bf16(&Kl[cur][(16 + lo) * 64 + 32 + 16 * half]);
            s1 = wmma_bf16(A1, Bk, s1);
        }
        int mv0 = mrow[k0 + lo];
        int mv1 = mrow[k0 + 16 + lo];

#pragma unroll
        for (int i = 0; i < 8; ++i) {
            float a = (mv0 == 0) ? -1e20f : s0[i] * SC;
            float c = (mv1 == 0) ? -1e20f : s1[i] * SC;
            float mx = fmaxf(a, c);
#pragma unroll
            for (int d = 1; d < 16; d <<= 1) mx = fmaxf(mx, __shfl_xor(mx, d, 16));
            float mnew  = fmaxf(rm[i], mx);
            float alpha = exp2f((rm[i] - mnew) * LOG2E);
            rm[i] = mnew;
            float p0 = exp2f((a - mnew) * LOG2E);
            float p1 = exp2f((c - mnew) * LOG2E);
            float ps = p0 + p1;
#pragma unroll
            for (int d = 1; d < 16; d <<= 1) ps += __shfl_xor(ps, d, 16);
            rl[i] = rl[i] * alpha + ps;
            o0[i] *= alpha; o1[i] *= alpha; o2[i] *= alpha; o3[i] *= alpha;
            // P tile to LDS: row M = i + 8*half, cols lo and 16+lo (row stride 32)
            Pl[wave][(i + 8 * half) * 32 + lo]      = (bf16_t)p0;
            Pl[wave][(i + 8 * half) * 32 + 16 + lo] = (bf16_t)p1;
        }
        asm volatile("s_wait_dscnt 0" ::: "memory");

        // Re-read P as A-fragment (16x32), then O += P @ V (4 d-tiles) from LDS
        v16bf Pa = load_a_bf16(&Pl[wave][lo * 32], half);
        v16bf Bv;
        Bv = load_b_bf16(&Vl[cur][(0  + lo) * 32 + 16 * half]);
        o0 = wmma_bf16(Pa, Bv, o0);
        Bv = load_b_bf16(&Vl[cur][(16 + lo) * 32 + 16 * half]);
        o1 = wmma_bf16(Pa, Bv, o1);
        Bv = load_b_bf16(&Vl[cur][(32 + lo) * 32 + 16 * half]);
        o2 = wmma_bf16(Pa, Bv, o2);
        Bv = load_b_bf16(&Vl[cur][(48 + lo) * 32 + 16 * half]);
        o3 = wmma_bf16(Pa, Bv, o3);

        __syncthreads();   // buffer `cur` may be refilled next iteration
    }

    // Normalize and store attention output row-major [r, 1024] (bf16)
#pragma unroll
    for (int i = 0; i < 8; ++i) {
        float inv = 1.0f / rl[i];
        bf16_t* orow = ob + (size_t)(n * SEQL + q0 + i + 8 * half) * ED + h * HD;
        orow[0  + lo] = (bf16_t)(o0[i] * inv);
        orow[16 + lo] = (bf16_t)(o1[i] * inv);
        orow[32 + lo] = (bf16_t)(o2[i] * inv);
        orow[48 + lo] = (bf16_t)(o3[i] * inv);
    }
}

// ---------------------------------------------------------------------------
// Kernel 3: out = ob @ Wo^T + bo  (8192 x 1024 x 1024 bf16 WMMA GEMM, f32 out)
// One wave = 16x64 output tile, K loop of 32 -> 128 WMMAs/wave.
// ---------------------------------------------------------------------------
__global__ void out_proj(const bf16_t* __restrict__ ob, const bf16_t* __restrict__ Wob,
                         const float* __restrict__ bo, float* __restrict__ out) {
    int wave = threadIdx.x >> 5, lane = threadIdx.x & 31;
    int lo = lane & 15, half = lane >> 4;
    int g  = blockIdx.x * 8 + wave;   // 8192 waves
    int r0 = (g >> 4) * 16;
    int e0 = (g & 15) * 64;

    v8f c0 = {}, c1 = {}, c2 = {}, c3 = {};
    const bf16_t* arow = ob + (size_t)(r0 + lo) * ED;

    for (int f0 = 0; f0 < ED; f0 += 32) {
        v16bf A = load_a_bf16(arow + f0, half);
        v16bf B;
        B  = load_b_bf16(Wob + (size_t)(e0 + lo)      * ED + f0 + 16 * half);
        c0 = wmma_bf16(A, B, c0);
        B  = load_b_bf16(Wob + (size_t)(e0 + 16 + lo) * ED + f0 + 16 * half);
        c1 = wmma_bf16(A, B, c1);
        B  = load_b_bf16(Wob + (size_t)(e0 + 32 + lo) * ED + f0 + 16 * half);
        c2 = wmma_bf16(A, B, c2);
        B  = load_b_bf16(Wob + (size_t)(e0 + 48 + lo) * ED + f0 + 16 * half);
        c3 = wmma_bf16(A, B, c3);
    }

    float b0 = bo[e0 + lo], b1 = bo[e0 + 16 + lo], b2 = bo[e0 + 32 + lo], b3 = bo[e0 + 48 + lo];
#pragma unroll
    for (int i = 0; i < 8; ++i) {
        float* orow = out + (size_t)(r0 + i + 8 * half) * ED + e0;
        orow[0  + lo] = c0[i] + b0;
        orow[16 + lo] = c1[i] + b1;
        orow[32 + lo] = c2[i] + b2;
        orow[48 + lo] = c3[i] + b3;
    }
}

extern "C" void kernel_launch(void* const* d_in, const int* in_sizes, int n_in,
                              void* d_out, int out_size, void* d_ws, size_t ws_size,
                              hipStream_t stream) {
    const float* values  = (const float*)d_in[0];
    const float* keys    = (const float*)d_in[1];
    const float* queries = (const float*)d_in[2];
    const int*   src_mask= (const int*)  d_in[3];
    const float* Wv      = (const float*)d_in[4];
    const float* Wk      = (const float*)d_in[5];
    const float* Wq      = (const float*)d_in[6];
    const float* Wo      = (const float*)d_in[7];
    const float* bo      = (const float*)d_in[8];
    float* out = (float*)d_out;

    const size_t PER = (size_t)NB * NH * SEQL * HD;   // 8M elems
    bf16_t* qb  = (bf16_t*)d_ws;
    bf16_t* kb  = qb  + PER;
    bf16_t* vtb = kb  + PER;
    bf16_t* ob  = vtb + PER;
    bf16_t* Wob = ob  + (size_t)NB * SEQL * ED;       // +8M -> +1M for Wo

    wo_to_bf16<<<(ED * ED) / 256, 256, 0, stream>>>(Wo, Wob);
    qkv_proj  <<<3072, 256, 0, stream>>>(values, keys, queries, Wv, Wk, Wq, qb, kb, vtb);
    flash_attn<<<1024, 256, 0, stream>>>(qb, kb, vtb, src_mask, ob);
    out_proj  <<<1024, 256, 0, stream>>>(ob, Wob, bo, out);
}
